// ScaledDotProductAttention_24026047054463
// MI455X (gfx1250) — matrix operand order repaired and checked by reading, not compiled, not run on hardware
//
#include <hip/hip_runtime.h>
#include <stdint.h>

// ---------------------------------------------------------------------------
// Scaled dot-product attention for MI455X (gfx1250, wave32, WMMA).
// B=4, H=16, S=1024, Dk=Dv=64.  Outputs: context [B,H,S,64] ++ attn [B,H,S,S].
// One workgroup = one (b*h, 32-row q-tile).  Full 32x1024 fp32 score panel
// lives in LDS (possible only with CDNA5's 320KB/WGP).  K/V tiles stream
// through a double-buffered LDS pair via GLOBAL_LOAD_ASYNC_TO_LDS_B128
// (ASYNCcnt), overlapping the copy engine with v_wmma compute.
// NOTE: no aggregate (array) locals may be derived from the dynamic-LDS
// symbol -- clang promotes them to global constant initializers containing
// addrspacecast(@smem), which ld.lld rejects.  Scalars only.
// ---------------------------------------------------------------------------

typedef __attribute__((ext_vector_type(16))) _Float16 v16h;
typedef __attribute__((ext_vector_type(8)))  float    v8f;

namespace cfg {
constexpr int B = 4, H = 16, S = 1024, D = 64;
constexpr int QT = 32;            // q rows per workgroup
constexpr int KT = 128;           // k rows per staged tile
constexpr int NKB = S / KT;       // 8 k-tiles
constexpr int SSTR  = 1025;       // score-panel row stride (f32), odd -> conflict-free
constexpr int KSTRF = 68;         // K/V tile row stride (f32): rows stagger 4 banks
constexpr int QSTR  = 66;         // Q tile row stride (f16)
constexpr float SCALE = 0.125f;   // 1/sqrt(64)
constexpr float NEG = -1e9f;

// dynamic-LDS carve (bytes)
constexpr int OFF_SCORES = 0;                                  // 32*1025*4 = 131200
constexpr int OFF_KV0  = OFF_SCORES + QT * SSTR * 4;           // 128*68*4  = 34816
constexpr int OFF_KV1  = OFF_KV0 + KT * KSTRF * 4;             // 34816
constexpr int OFF_Q    = OFF_KV1 + KT * KSTRF * 4;             // 32*66*2   = 4224
constexpr int OFF_PART = OFF_Q + QT * QSTR * 2;                // 32*8*4    = 1024
constexpr int OFF_RMAX = OFF_PART + QT * 8 * 4;                // 32*4
constexpr int OFF_RINV = OFF_RMAX + QT * 4;                    // 32*4
constexpr int LDS_BYTES = OFF_RINV + QT * 4;                   // = 206336
}

// -- CDNA5 async copy: memory -> LDS, no VGPR data, tracked by ASYNCcnt -----
__device__ __forceinline__ void async_b128(uint32_t ldsAddr, const float* g) {
  asm volatile("global_load_async_to_lds_b128 %0, %1, off"
               :: "v"(ldsAddr), "v"((uint64_t)(uintptr_t)g)
               : "memory");
}

template <int N>
__device__ __forceinline__ void wait_async() {
#if defined(__has_builtin) && __has_builtin(__builtin_amdgcn_s_wait_asynccnt)
  __builtin_amdgcn_s_wait_asynccnt((unsigned short)N);
#else
  asm volatile("s_wait_asynccnt %0" :: "i"(N) : "memory");
#endif
}

__global__ void __launch_bounds__(256)
ScaledDotProductAttention_24026047054463_kernel(
    const float* __restrict__ Q, const float* __restrict__ K,
    const float* __restrict__ V, const unsigned char* __restrict__ mask,
    float* __restrict__ ctx_out, float* __restrict__ attn_out)
{
  using namespace cfg;
  extern __shared__ char smem[];
  // scalars only -- see file header note
  float*    scores = reinterpret_cast<float*>(smem + OFF_SCORES);   // [QT][SSTR]
  float*    kv0    = reinterpret_cast<float*>(smem + OFF_KV0);      // [KT][KSTRF]
  float*    kv1    = reinterpret_cast<float*>(smem + OFF_KV1);      // [KT][KSTRF]
  _Float16* qlds   = reinterpret_cast<_Float16*>(smem + OFF_Q);     // [QT][QSTR]
  float*    part   = reinterpret_cast<float*>(smem + OFF_PART);     // [QT][8]
  float*    rmax   = reinterpret_cast<float*>(smem + OFF_RMAX);     // [QT]
  float*    rinv   = reinterpret_cast<float*>(smem + OFF_RINV);     // [QT]

  // low 32 bits of a generic shared pointer == LDS byte address
  const uint32_t ldsBase = (uint32_t)(uintptr_t)smem;
  const uint32_t kvA0 = ldsBase + OFF_KV0;
  const uint32_t kvA1 = ldsBase + OFF_KV1;

  const int tid  = threadIdx.x;
  const int lane = tid & 31;          // wave32 on gfx1250
  const int wave = tid >> 5;          // 8 waves
  const int n    = lane & 15;
  const bool hi  = lane >= 16;

  const int bh    = blockIdx.y;       // 0..63  (b*H + h)
  const int qbase = blockIdx.x * QT;  // 0..992

  const size_t headOff = (size_t)bh * S * D;

  // issue one 128x64 f32 tile (rows kt..kt+127 of K or V) as 8 async b128/lane
  auto issueTile = [&](const float* __restrict__ src, int kt, uint32_t bufAddr) {
#pragma unroll
    for (int i = 0; i < 8; ++i) {                 // 128*64*4B / 16B / 256thr = 8
      int e4 = tid + i * 256;
      int e = e4 * 4, row = e >> 6, d = e & 63;
      async_b128(bufAddr + (uint32_t)(row * KSTRF + d) * 4,
                 src + headOff + (size_t)kt * D + e);
    }
  };

  // ---------------- Phase 1: Q tile -> LDS (f16) --------------------------
  {
    const float4* g4 = reinterpret_cast<const float4*>(Q + headOff + (size_t)qbase * D);
#pragma unroll
    for (int i = 0; i < 2; ++i) {
      int e4 = tid + i * 256;
      float4 f = g4[e4];
      int e = e4 * 4, row = e >> 6, d = e & 63;
      _Float16* p = qlds + row * QSTR + d;
      p[0] = (_Float16)f.x; p[1] = (_Float16)f.y;
      p[2] = (_Float16)f.z; p[3] = (_Float16)f.w;
    }
  }

  // ---------------- Phase 2: S = (Q K^T) * scale, double-buffered ---------
  issueTile(K, 0, kvA0);
  for (int kb = 0; kb < NKB; ++kb) {
    const int cur = kb & 1;
    if (kb + 1 < NKB) {                            // prefetch next tile
      issueTile(K, (kb + 1) * KT, cur ? kvA0 : kvA1);
      wait_async<8>();                             // oldest 8 (current tile) done
    } else {
      wait_async<0>();
    }
    __syncthreads();                               // all waves' copies landed

    const float* kv = cur ? kv1 : kv0;
    const int kt = kb * KT;
#pragma unroll
    for (int trow = 0; trow < 2; ++trow) {
      v8f c = {};
#pragma unroll
      for (int kc = 0; kc < 2; ++kc) {             // Dk = 64 = 2 x 32
        v16h a, b;
        const int arow = trow * 16 + n;            // A: M = lane%16
#pragma unroll
        for (int h = 0; h < 16; ++h) {
          int kd = kc * 32 + (hi ? 16 : 0) + h;
          a[h] = qlds[arow * QSTR + kd];
        }
        const int key = wave * 16 + n;             // B: N = lane%16 (key index)
#pragma unroll
        for (int h = 0; h < 16; ++h) {             // ds_load_b128 + v_cvt_pk_f16_f32
          int kd = kc * 32 + (hi ? 16 : 0) + h;
          b[h] = (_Float16)kv[key * KSTRF + kd];
        }
        c = __builtin_amdgcn_wmma_f32_16x16x32_f16(false, a, false, b,
                                                   (short)0, c, false, false);
      }
#pragma unroll
      for (int r = 0; r < 8; ++r) {
        int M = trow * 16 + r + (hi ? 8 : 0);
        scores[M * SSTR + kt + wave * 16 + n] = c[r] * SCALE;
      }
    }
    __syncthreads();                               // buffer reusable in kb+2
  }

  // ---------------- Phase 3: masked softmax over LDS panel ----------------
  const int row = tid >> 3;
  const int seg = tid & 7;
  float* srow = scores + row * SSTR + seg * 128;
  const size_t abase = ((size_t)bh * S + qbase + row) * S + seg * 128;
  const uint32_t* m32 = reinterpret_cast<const uint32_t*>(mask + abase);

  float lmax = -3.4e38f;
#pragma unroll 4
  for (int cc = 0; cc < 32; ++cc) {
    uint32_t mw = m32[cc];
#pragma unroll
    for (int j = 0; j < 4; ++j) {
      float s = srow[cc * 4 + j];
      if ((mw >> (8 * j)) & 0xFF) s = NEG;         // mask==true -> -1e9
      srow[cc * 4 + j] = s;
      lmax = fmaxf(lmax, s);
    }
  }
  part[row * 8 + seg] = lmax;
  __syncthreads();
  if (tid < QT) {
    float m = -3.4e38f;
#pragma unroll
    for (int j = 0; j < 8; ++j) m = fmaxf(m, part[tid * 8 + j]);
    rmax[tid] = m;
  }
  __syncthreads();

  const float rm = rmax[row];
  float lsum = 0.f;
#pragma unroll 4
  for (int c = 0; c < 128; ++c) {
    float p = __expf(srow[c] - rm);
    srow[c] = p;
    lsum += p;
  }
  part[row * 8 + seg] = lsum;
  __syncthreads();
  if (tid < QT) {
    float s = 0.f;
#pragma unroll
    for (int j = 0; j < 8; ++j) s += part[tid * 8 + j];
    rinv[tid] = 1.0f / s;
  }
  __syncthreads();

  {
    const float inv = rinv[row];
    float4* ao = reinterpret_cast<float4*>(attn_out + abase);
#pragma unroll 4
    for (int cc = 0; cc < 32; ++cc) {
      float4 o;
      o.x = srow[cc * 4 + 0] * inv; o.y = srow[cc * 4 + 1] * inv;
      o.z = srow[cc * 4 + 2] * inv; o.w = srow[cc * 4 + 3] * inv;
      ao[cc] = o;
    }
  }

  // ---------------- Phase 4: context = P V, double-buffered ---------------
  const int mrow = wave >> 2, ncol = wave & 3;     // 2x4 tiles of 16x16 over 32x64
  const int qrow = mrow * 16 + n;
  v8f acc = {};
  issueTile(V, 0, kvA0);
  for (int kb = 0; kb < NKB; ++kb) {
    const int cur = kb & 1;
    if (kb + 1 < NKB) {
      issueTile(V, (kb + 1) * KT, cur ? kvA0 : kvA1);
      wait_async<8>();
    } else {
      wait_async<0>();
    }
    __syncthreads();

    const float* kv = cur ? kv1 : kv0;
    const float inv = rinv[qrow];                  // fold softmax normalization
#pragma unroll
    for (int chunk = 0; chunk < 4; ++chunk) {      // 128 = 4 x 32
      v16h a, b;
#pragma unroll
      for (int h = 0; h < 16; ++h) {
        int kd = (hi ? 16 : 0) + h;
        int col = kb * KT + chunk * 32 + kd;
        a[h] = (_Float16)(scores[qrow * SSTR + col] * inv);
      }
#pragma unroll
      for (int h = 0; h < 16; ++h) {
        int vrow = chunk * 32 + (hi ? 16 : 0) + h;
        b[h] = (_Float16)kv[vrow * KSTRF + ncol * 16 + n];
      }
      acc = __builtin_amdgcn_wmma_f32_16x16x32_f16(false, a, false, b,
                                                   (short)0, acc, false, false);
    }
    __syncthreads();
  }
#pragma unroll
  for (int r = 0; r < 8; ++r) {
    int q = qbase + mrow * 16 + r + (hi ? 8 : 0);
    ctx_out[((size_t)bh * S + q) * D + ncol * 16 + n] = acc[r];
  }
}

extern "C" void kernel_launch(void* const* d_in, const int* in_sizes, int n_in,
                              void* d_out, int out_size, void* d_ws, size_t ws_size,
                              hipStream_t stream) {
  using namespace cfg;
  (void)in_sizes; (void)n_in; (void)out_size; (void)d_ws; (void)ws_size;
  const float* Q = (const float*)d_in[0];
  const float* K = (const float*)d_in[1];
  const float* V = (const float*)d_in[2];
  const unsigned char* mask = (const unsigned char*)d_in[3];

  float* ctx  = (float*)d_out;                                  // [B,H,S,64]
  float* attn = ctx + (size_t)B * H * S * D;                    // [B,H,S,S]

  static_assert(LDS_BYTES == 206336, "LDS carve changed");
  (void)hipFuncSetAttribute(
      (const void*)ScaledDotProductAttention_24026047054463_kernel,
      hipFuncAttributeMaxDynamicSharedMemorySize, LDS_BYTES);

  dim3 grid(S / QT, B * H);   // (32, 64)
  dim3 block(256);            // 8 wave32 waves
  ScaledDotProductAttention_24026047054463_kernel
      <<<grid, block, LDS_BYTES, stream>>>(Q, K, V, mask, ctx, attn);
}
